// MultiHeadAttention_82669530513713
// MI455X (gfx1250) — compile-verified, matrix operand order checked by reading, not run on hardware
//
#include <hip/hip_runtime.h>

typedef __attribute__((ext_vector_type(16))) _Float16 v16h;
typedef __attribute__((ext_vector_type(8)))  _Float16 v8h;
typedef __attribute__((ext_vector_type(8)))  float    v8f;
typedef int v4i __attribute__((vector_size(16)));

#define B_   2
#define S_   4096
#define E_   512
#define H_   8
#define D_   64
#define BS_  (B_ * S_)     /* 8192 */
#define N3E  (3 * E_)      /* 1536 */

// ---------------------------------------------------------------------------
// Async global->LDS staging (GLOBAL_LOAD_ASYNC_TO_LDS_B128, ASYNCcnt) when the
// toolchain exposes the builtin; otherwise fall back to load+ds_store.
// Probe-learned signature: (int4 AS1*, int4 AS3*, imm offset, imm cpol).
// ---------------------------------------------------------------------------
#if defined(__has_builtin)
#if __has_builtin(__builtin_amdgcn_global_load_async_to_lds_b128)
#define HAVE_ASYNC_LDS 1
#endif
#endif

__device__ __forceinline__ void stage_b128(const _Float16* g, _Float16* l) {
#if defined(HAVE_ASYNC_LDS)
  void* gv = (void*)g;  // drop const, then addrspace-qualify
  __builtin_amdgcn_global_load_async_to_lds_b128(
      (__attribute__((address_space(1))) v4i*)gv,
      (__attribute__((address_space(3))) v4i*)l, 0, 0);
#else
  *(v8h*)l = *(const v8h*)g;
#endif
}

__device__ __forceinline__ void async_join() {
#if defined(HAVE_ASYNC_LDS)
#if __has_builtin(__builtin_amdgcn_s_wait_asynccnt)
  __builtin_amdgcn_s_wait_asynccnt(0);
#else
  asm volatile("s_wait_asynccnt 0" ::: "memory");
#endif
#endif
}

// ---------------------------------------------------------------------------
// Fragment loader for 16-bit A/B operands of v_wmma_f32_16x16x32_f16.
// ISA layout (16-bit A 16x32): lane l holds row r = l&15; kh = l>>4 selects
// K runs {kh*8 .. kh*8+7} and {16+kh*8 .. 16+kh*8+7}. B uses the same packing
// when the B matrix is stored [N][K] row-major (i.e. column-major B).
// Requires: tile 16B-aligned, stride multiple of 8 halves.
// ---------------------------------------------------------------------------
__device__ __forceinline__ v16h frag_ld(const _Float16* tile, int stride, int lane) {
  const int r  = lane & 15;
  const int kh = lane >> 4;
  const v8h lo = *(const v8h*)(tile + r * stride + kh * 8);
  const v8h hi = *(const v8h*)(tile + r * stride + kh * 8 + 16);
  v16h out;
#pragma unroll
  for (int i = 0; i < 8; ++i) { out[i] = lo[i]; out[8 + i] = hi[i]; }
  return out;
}

__device__ __forceinline__ v8f wmma16(v16h a, v16h b, v8f c) {
  // D = A(16x32) * B(32x16) + C, f32 accumulate
  return __builtin_amdgcn_wmma_f32_16x16x32_f16(false, a, false, b, (short)0, c,
                                                false, false);
}

// ---------------------------------------------------------------------------
// Prep kernels: f32 -> f16 convert, and weight transpose to [N][K=512] f16.
// ---------------------------------------------------------------------------
__global__ void k_cvt_f32_f16(const float* __restrict__ in,
                              _Float16* __restrict__ out, int n) {
  const int i = blockIdx.x * blockDim.x + threadIdx.x;
  if (i < n) out[i] = (_Float16)in[i];
}

__global__ void k_transpose_w(const float* __restrict__ W,   // [512][N]
                              _Float16* __restrict__ Wt,     // [N][512]
                              int N) {
  const int i = blockIdx.x * blockDim.x + threadIdx.x;       // over N*512
  const int n = i >> 9;
  const int k = i & 511;
  if (n < N) Wt[(size_t)n * 512 + k] = (_Float16)W[(size_t)k * N + n];
}

// ---------------------------------------------------------------------------
// GEMM1: qkv = Xh[8192,512] @ Wqkv + b  -> scatter f16 Q/K/V as [b,h,s,d].
// Block 256 thr = 8 waves; block tile 128x64; wave tile 32x32 (2x2 frags).
// LDS row stride 40 halves: 16B-aligned b128 frag reads, conflict-free.
// ---------------------------------------------------------------------------
__global__ __launch_bounds__(256) void k_gemm_qkv(
    const _Float16* __restrict__ Xh, const _Float16* __restrict__ Wt,
    const float* __restrict__ bias, _Float16* __restrict__ Q,
    _Float16* __restrict__ K, _Float16* __restrict__ V) {
  __shared__ _Float16 Al[128 * 40];
  __shared__ _Float16 Bl[64 * 40];
  const int tid = threadIdx.x, lane = tid & 31, wave = tid >> 5;
  const int mblk = blockIdx.x * 128, nblk = blockIdx.y * 64;
  const int wm = (wave & 3) * 32, wn = (wave >> 2) * 32;
  const int ar = tid >> 1, ac = (tid & 1) * 16;   // A stage: 128 rows x 32 halves
  const int bn = tid >> 2, bc = (tid & 3) * 8;    // B stage: 64 rows x 32 halves
  v8f a00 = {}, a01 = {}, a10 = {}, a11 = {};

  for (int kb = 0; kb < E_; kb += 32) {
    __syncthreads();
    const _Float16* ap = Xh + (size_t)(mblk + ar) * E_ + kb + ac;
    stage_b128(ap,     &Al[ar * 40 + ac]);
    stage_b128(ap + 8, &Al[ar * 40 + ac + 8]);
    stage_b128(Wt + (size_t)(nblk + bn) * E_ + kb + bc, &Bl[bn * 40 + bc]);
    if (kb + 32 < E_) __builtin_prefetch(ap + 32, 0, 2);  // global_prefetch_b8
    async_join();
    __syncthreads();
    const v16h fa0 = frag_ld(&Al[wm * 40], 40, lane);
    const v16h fa1 = frag_ld(&Al[(wm + 16) * 40], 40, lane);
    const v16h fb0 = frag_ld(&Bl[wn * 40], 40, lane);
    const v16h fb1 = frag_ld(&Bl[(wn + 16) * 40], 40, lane);
    a00 = wmma16(fa0, fb0, a00); a01 = wmma16(fa0, fb1, a01);
    a10 = wmma16(fa1, fb0, a10); a11 = wmma16(fa1, fb1, a11);
  }

  const int half = lane >> 4, nn = lane & 15;
#pragma unroll
  for (int fm = 0; fm < 2; ++fm) {
#pragma unroll
    for (int fn = 0; fn < 2; ++fn) {
      const v8f acc = fm == 0 ? (fn == 0 ? a00 : a01) : (fn == 0 ? a10 : a11);
      const int col = nblk + wn + fn * 16 + nn;
      const float bv = bias[col];
      const int sel = col >> 9, cc = col & 511, h = cc >> 6, d = cc & 63;
      _Float16* dst = (sel == 0) ? Q : ((sel == 1) ? K : V);
#pragma unroll
      for (int i = 0; i < 8; ++i) {
        const int row = mblk + wm + fm * 16 + half * 8 + i;  // token index
        const int bb = row >> 12, ss = row & 4095;
        dst[((size_t)(bb * H_ + h) * S_ + ss) * D_ + d] = (_Float16)(acc[i] + bv);
      }
    }
  }
}

// ---------------------------------------------------------------------------
// Flash attention: block = 128 thr = 4 waves, each wave owns 16 queries
// (block covers 64 queries of one (b,h)). Streams 32 keys per iteration.
// Online softmax in log2 domain; P bounced through per-wave LDS to convert
// the D-fragment layout into an A-fragment for P·V.
// All of K+V per head is ~1 MB f16 -> resident in the 192 MB L2, so this
// phase is WMMA-bound, not HBM-bound.
// ---------------------------------------------------------------------------
__global__ __launch_bounds__(128) void k_flash(const _Float16* __restrict__ Qh,
                                               const _Float16* __restrict__ Kh,
                                               const _Float16* __restrict__ Vh,
                                               _Float16* __restrict__ AO) {
  __shared__ _Float16 Kl[32 * 72];      // [key][d], stride 72: conflict-free
  __shared__ _Float16 Vt[64 * 40];      // [d][key], stride 40: conflict-free
  __shared__ _Float16 Pl[4][16 * 40];   // per-wave P tile [16 q][32 key]

  const int tid = threadIdx.x, lane = tid & 31, wave = tid >> 5;
  const int bh = blockIdx.y;                       // b*8 + h
  const int qbase = blockIdx.x * 64 + wave * 16;

  const _Float16* Qp = Qh + ((size_t)bh * S_ + qbase) * D_;
  const v16h qf0 = frag_ld(Qp, D_, lane);          // d 0..31
  const v16h qf1 = frag_ld(Qp + 32, D_, lane);     // d 32..63
  const _Float16* Kp = Kh + (size_t)bh * S_ * D_;
  const _Float16* Vp = Vh + (size_t)bh * S_ * D_;

  v8f o0 = {}, o1 = {}, o2 = {}, o3 = {};
  float mrow[8], lrow[8];
#pragma unroll
  for (int i = 0; i < 8; ++i) { mrow[i] = -1e30f; lrow[i] = 0.f; }

  // (1/sqrt(E)) * log2(e): do softmax in base-2 so exp lowers to v_exp_f32
  const float kScale = 0.044194173824159216f * 1.4426950408889634f;
  const int skey = tid >> 2, schunk = tid & 3;

  for (int kb = 0; kb < S_; kb += 32) {
    __syncthreads();  // protect previous iteration's K/V reads
    {
      const _Float16* krow = Kp + (size_t)(kb + skey) * D_ + schunk * 16;
      stage_b128(krow,     &Kl[skey * 72 + schunk * 16]);
      stage_b128(krow + 8, &Kl[skey * 72 + schunk * 16 + 8]);
      const v16h vv = *(const v16h*)(Vp + (size_t)(kb + skey) * D_ + schunk * 16);
#pragma unroll
      for (int j = 0; j < 16; ++j) Vt[(schunk * 16 + j) * 40 + skey] = vv[j];
    }
    async_join();
    __syncthreads();

    // scores: S[16 q][32 key] as two 16x16 f32 D-fragments
    v8f s0 = {}, s1 = {};
    {
      const v16h b00 = frag_ld(&Kl[0], 72, lane);
      const v16h b01 = frag_ld(&Kl[32], 72, lane);
      s0 = wmma16(qf0, b00, s0); s0 = wmma16(qf1, b01, s0);
      const v16h b10 = frag_ld(&Kl[16 * 72], 72, lane);
      const v16h b11 = frag_ld(&Kl[16 * 72 + 32], 72, lane);
      s1 = wmma16(qf0, b10, s1); s1 = wmma16(qf1, b11, s1);
    }

    float t0[8], t1[8], mc[8];
#pragma unroll
    for (int i = 0; i < 8; ++i) {
      t0[i] = s0[i] * kScale; t1[i] = s1[i] * kScale;
      mc[i] = fmaxf(t0[i], t1[i]);
    }
    // row max over 16 lanes of the half-wave (row m = (lane>>4)*8 + i)
#pragma unroll
    for (int off = 8; off >= 1; off >>= 1)
#pragma unroll
      for (int i = 0; i < 8; ++i) mc[i] = fmaxf(mc[i], __shfl_xor(mc[i], off));

    float al[8], rs[8];
#pragma unroll
    for (int i = 0; i < 8; ++i) {
      const float mn = fmaxf(mrow[i], mc[i]);
      al[i] = __builtin_amdgcn_exp2f(mrow[i] - mn);
      t0[i] = __builtin_amdgcn_exp2f(t0[i] - mn);
      t1[i] = __builtin_amdgcn_exp2f(t1[i] - mn);
      rs[i] = t0[i] + t1[i];
      mrow[i] = mn;
    }
#pragma unroll
    for (int off = 8; off >= 1; off >>= 1)
#pragma unroll
      for (int i = 0; i < 8; ++i) rs[i] += __shfl_xor(rs[i], off);
#pragma unroll
    for (int i = 0; i < 8; ++i) {
      lrow[i] = lrow[i] * al[i] + rs[i];
      o0[i] *= al[i]; o1[i] *= al[i]; o2[i] *= al[i]; o3[i] *= al[i];
    }

    // D-layout P -> per-wave LDS -> A-fragment (LDS is in-order per wave)
    {
      _Float16* Pw = Pl[wave];
      const int half = lane >> 4, nn = lane & 15;
#pragma unroll
      for (int i = 0; i < 8; ++i) {
        Pw[(half * 8 + i) * 40 + nn]      = (_Float16)t0[i];
        Pw[(half * 8 + i) * 40 + 16 + nn] = (_Float16)t1[i];
      }
      asm volatile("s_wait_dscnt 0" ::: "memory");
      const v16h pa = frag_ld(Pw, 40, lane);
      o0 = wmma16(pa, frag_ld(&Vt[0], 40, lane), o0);
      o1 = wmma16(pa, frag_ld(&Vt[16 * 40], 40, lane), o1);
      o2 = wmma16(pa, frag_ld(&Vt[32 * 40], 40, lane), o2);
      o3 = wmma16(pa, frag_ld(&Vt[48 * 40], 40, lane), o3);
    }
  }

  // normalize and merge heads: AO[b*S+s][h*64 + d] f16
  const int b = bh >> 3, h = bh & 7;
  const int half = lane >> 4, nn = lane & 15;
  float inv[8];
#pragma unroll
  for (int i = 0; i < 8; ++i) inv[i] = 1.0f / lrow[i];
#pragma unroll
  for (int i = 0; i < 8; ++i) {
    const size_t row = (size_t)b * S_ + qbase + half * 8 + i;
    _Float16* dst = AO + row * E_ + h * D_;
    dst[nn]      = (_Float16)(o0[i] * inv[i]);
    dst[16 + nn] = (_Float16)(o1[i] * inv[i]);
    dst[32 + nn] = (_Float16)(o2[i] * inv[i]);
    dst[48 + nn] = (_Float16)(o3[i] * inv[i]);
  }
}

// ---------------------------------------------------------------------------
// GEMM3: out = AO[8192,512] @ W_out + b_out (f32 result)
// ---------------------------------------------------------------------------
__global__ __launch_bounds__(256) void k_gemm_out(
    const _Float16* __restrict__ AO, const _Float16* __restrict__ Wt,
    const float* __restrict__ bias, float* __restrict__ out) {
  __shared__ _Float16 Al[128 * 40];
  __shared__ _Float16 Bl[64 * 40];
  const int tid = threadIdx.x, lane = tid & 31, wave = tid >> 5;
  const int mblk = blockIdx.x * 128, nblk = blockIdx.y * 64;
  const int wm = (wave & 3) * 32, wn = (wave >> 2) * 32;
  const int ar = tid >> 1, ac = (tid & 1) * 16;
  const int bn = tid >> 2, bc = (tid & 3) * 8;
  v8f a00 = {}, a01 = {}, a10 = {}, a11 = {};

  for (int kb = 0; kb < E_; kb += 32) {
    __syncthreads();
    const _Float16* ap = AO + (size_t)(mblk + ar) * E_ + kb + ac;
    stage_b128(ap,     &Al[ar * 40 + ac]);
    stage_b128(ap + 8, &Al[ar * 40 + ac + 8]);
    stage_b128(Wt + (size_t)(nblk + bn) * E_ + kb + bc, &Bl[bn * 40 + bc]);
    if (kb + 32 < E_) __builtin_prefetch(ap + 32, 0, 2);
    async_join();
    __syncthreads();
    const v16h fa0 = frag_ld(&Al[wm * 40], 40, lane);
    const v16h fa1 = frag_ld(&Al[(wm + 16) * 40], 40, lane);
    const v16h fb0 = frag_ld(&Bl[wn * 40], 40, lane);
    const v16h fb1 = frag_ld(&Bl[(wn + 16) * 40], 40, lane);
    a00 = wmma16(fa0, fb0, a00); a01 = wmma16(fa0, fb1, a01);
    a10 = wmma16(fa1, fb0, a10); a11 = wmma16(fa1, fb1, a11);
  }

  const int half = lane >> 4, nn = lane & 15;
#pragma unroll
  for (int fm = 0; fm < 2; ++fm) {
#pragma unroll
    for (int fn = 0; fn < 2; ++fn) {
      const v8f acc = fm == 0 ? (fn == 0 ? a00 : a01) : (fn == 0 ? a10 : a11);
      const int col = nblk + wn + fn * 16 + nn;
      const float bv = bias[col];
#pragma unroll
      for (int i = 0; i < 8; ++i) {
        const int row = mblk + wm + fm * 16 + half * 8 + i;
        out[(size_t)row * E_ + col] = acc[i] + bv;
      }
    }
  }
}

// ---------------------------------------------------------------------------
// Workspace layout (bytes, all 256B-aligned). Total ~42 MiB.
// ---------------------------------------------------------------------------
#define OFF_XH     ((size_t)0)                         // 8192*512*2  = 8 MiB
#define OFF_WQKVT  ((size_t)8388608)                   // 1536*512*2  = 1.5 MiB
#define OFF_WOUTT  ((size_t)9961472)                   // 512*512*2   = 0.5 MiB
#define OFF_Q      ((size_t)10485760)                  // 8 MiB
#define OFF_K      ((size_t)18874368)                  // 8 MiB
#define OFF_V      ((size_t)27262976)                  // 8 MiB
#define OFF_AO     ((size_t)35651584)                  // 8 MiB (end 42 MiB)

extern "C" void kernel_launch(void* const* d_in, const int* in_sizes, int n_in,
                              void* d_out, int out_size, void* d_ws,
                              size_t ws_size, hipStream_t stream) {
  (void)in_sizes; (void)n_in; (void)out_size; (void)ws_size;
  const float* x     = (const float*)d_in[0];
  const float* Wqkv  = (const float*)d_in[1];
  const float* bqkv  = (const float*)d_in[2];
  const float* Wout  = (const float*)d_in[3];
  const float* bout  = (const float*)d_in[4];
  float* out = (float*)d_out;

  char* ws = (char*)d_ws;
  _Float16* Xh    = (_Float16*)(ws + OFF_XH);
  _Float16* WqkvT = (_Float16*)(ws + OFF_WQKVT);
  _Float16* WoutT = (_Float16*)(ws + OFF_WOUTT);
  _Float16* Qh    = (_Float16*)(ws + OFF_Q);
  _Float16* Kh    = (_Float16*)(ws + OFF_K);
  _Float16* Vh    = (_Float16*)(ws + OFF_V);
  _Float16* AO    = (_Float16*)(ws + OFF_AO);

  // 1) precision conversion / weight transposes (tiny, bandwidth-bound)
  k_cvt_f32_f16<<<(BS_ * E_) / 256, 256, 0, stream>>>(x, Xh, BS_ * E_);
  k_transpose_w<<<(N3E * 512) / 256, 256, 0, stream>>>(Wqkv, WqkvT, N3E);
  k_transpose_w<<<(E_ * 512) / 256, 256, 0, stream>>>(Wout, WoutT, E_);

  // 2) fused QKV projection (WMMA) -> Q/K/V f16 [b,h,s,d]
  k_gemm_qkv<<<dim3(BS_ / 128, N3E / 64), 256, 0, stream>>>(Xh, WqkvT, bqkv,
                                                            Qh, Kh, Vh);
  // 3) flash attention (WMMA QK^T + online softmax + WMMA PV)
  k_flash<<<dim3(S_ / 64, B_ * H_), 128, 0, stream>>>(Qh, Kh, Vh, AO);

  // 4) output projection (WMMA) -> f32
  k_gemm_out<<<dim3(BS_ / 128, E_ / 64), 256, 0, stream>>>(AO, WoutT, bout, out);
}
